// ProjNorm_82386062672007
// MI455X (gfx1250) — matrix operand order repaired and checked by reading, not compiled, not run on hardware
//
#include <hip/hip_runtime.h>
#include <hip/hip_bf16.h>
#include <cstdint>

// ============================================================================
// ProjNorm loss on MI455X (gfx1250).
//   Phase 1: O(N^2) reductions + one WMMA GEMM (cov@cov) -> cov_y (= c1).
//   Phase 2: Bures-Wasserstein tr(sqrtm(c1*c2)) via coupled Newton-Schulz
//            (pure GEMMs on v_wmma_f32_16x16x32_bf16).
// Matrices stored as split-bf16 PLANES (hi + residual lo), DEINTERLEAVED so
// WMMA fragments are direct 32-bit LDS loads (no v_mov_b16 marshalling).
// GEMM is "NT": right operand stored K-major (transposed); producers emit
// transposed copies (symmetric inputs get them for free).
// ============================================================================

#define GN 2048
#define NNE (GN * GN)
#define NS_ITERS 16

typedef __attribute__((ext_vector_type(16))) __bf16 bf16x16;
typedef __attribute__((ext_vector_type(8)))  float  f32x8;
typedef __attribute__((ext_vector_type(4)))  int    v4i_t;

// ---------------- split-bf16 helpers ----------------
__device__ __forceinline__ uint32_t f2bf_bits(float x) {
  uint32_t u = __float_as_uint(x);
  return (u + 0x7FFFu + ((u >> 16) & 1u)) >> 16;   // round-to-nearest-even
}
__device__ __forceinline__ float bfbits2f(uint32_t b) { return __uint_as_float(b << 16); }
__device__ __forceinline__ __bf16 bf_bits(uint32_t b) {
  return __builtin_bit_cast(__bf16, (unsigned short)(b & 0xFFFFu));
}
__device__ __forceinline__ void split2(float x, __bf16& h, __bf16& l) {
  uint32_t hb = f2bf_bits(x);
  h = bf_bits(hb);
  l = bf_bits(f2bf_bits(x - bfbits2f(hb)));
}
__device__ __forceinline__ float up2(__bf16 h, __bf16 l) {
  return (float)h + (float)l;
}

// ---------------- async LDS staging (guarded; fallback = plain copy) --------
#if defined(__AMDGCN__) && __has_builtin(__builtin_amdgcn_global_load_async_to_lds_b128)
#define HAS_ASYNC 1
#define AS1 __attribute__((address_space(1)))
#define AS3 __attribute__((address_space(3)))
#if __has_builtin(__builtin_amdgcn_s_wait_asynccnt)
#define WAIT_ASYNC(n) __builtin_amdgcn_s_wait_asynccnt(n)
#else
#define WAIT_ASYNC(n) asm volatile("s_wait_asynccnt %0" ::"n"(n) : "memory")
#endif
#else
#define HAS_ASYNC 0
#define WAIT_ASYNC(n) ((void)0)
#endif

// ---------------- block reduction helper ----------------
__device__ __forceinline__ float block_sum(float v) {
  __shared__ float sb[256];
  const int t = threadIdx.x;
  __syncthreads();
  sb[t] = v;
  __syncthreads();
  #pragma unroll
  for (int s = 128; s > 0; s >>= 1) {
    if (t < s) sb[t] += sb[t + s];
    __syncthreads();
  }
  return sb[0];
}

// ============================================================================
// NT-GEMM: C[m][n] = sum_k A[m][k] * Bt[n][k]   (all operands bf16 planes)
// Effective product: C = A @ B where Bt is B stored transposed.
// Split precision: C ~= Ah@Bh + Ah@Bl + Al@Bh (f32 accumulation).
// Block tile 128x128, K-step 32; 8 waves; wave tile 32x64 (2x4 WMMA tiles).
// ============================================================================
#define BLKT 128
#define KC 32
#define LDT 40   // bf16 row stride of LDS tiles: 80 B (16B-aligned, bank-spread)

union frag_u {
  bf16x16 v;
  uint32_t w[8];
};

__global__ __launch_bounds__(256) void k_gemm_nt(
    const __bf16* __restrict__ Ah, const __bf16* __restrict__ Al,
    const __bf16* __restrict__ Bh, const __bf16* __restrict__ Bl,  // K-major (transposed)
    __bf16* __restrict__ Ch, __bf16* __restrict__ Cl,
    __bf16* __restrict__ Cth, __bf16* __restrict__ Ctl) {          // optional transposed out
  __shared__ __bf16 shAh[2][BLKT * LDT];
  __shared__ __bf16 shAl[2][BLKT * LDT];
  __shared__ __bf16 shBh[2][BLKT * LDT];
  __shared__ __bf16 shBl[2][BLKT * LDT];

  const int t    = threadIdx.x;
  const int lane = t & 31;
  const int wid  = t >> 5;
  const int wm   = (wid & 3) * 32;   // wave row offset inside block tile
  const int wn   = (wid >> 2) * 64;  // wave col offset inside block tile
  const int hlf  = lane >> 4;
  const int l16  = lane & 15;
  const int rowBase = blockIdx.y * BLKT;
  const int colBase = blockIdx.x * BLKT;

  f32x8 acc[2][4];
  #pragma unroll
  for (int mt = 0; mt < 2; ++mt)
    #pragma unroll
    for (int nt = 0; nt < 4; ++nt)
      acc[mt][nt] = (f32x8){0.f, 0.f, 0.f, 0.f, 0.f, 0.f, 0.f, 0.f};

  // stage one K-slice: 4 tiles of 128x32 bf16 (8 KB each), 16B segments
  auto stage = [&](int buf, int kt) {
    const int k0 = kt * KC;
    #pragma unroll
    for (int p = 0; p < 2; ++p) {
      const int s = t + p * 256;        // 512 segments per tile
      const int r = s >> 2;             // tile row
      const int c = (s & 3) * 8;        // bf16 offset within row (8 bf16 = 16B)
      const size_t goff = (size_t)r * GN + (k0 + c);
      const int     loff = r * LDT + c;
      const __bf16* gah = Ah + (size_t)(rowBase)*GN + goff;
      const __bf16* gal = Al + (size_t)(rowBase)*GN + goff;
      const __bf16* gbh = Bh + (size_t)(colBase)*GN + goff;
      const __bf16* gbl = Bl + (size_t)(colBase)*GN + goff;
#if HAS_ASYNC
      __builtin_amdgcn_global_load_async_to_lds_b128((AS1 v4i_t*)gah, (AS3 v4i_t*)&shAh[buf][loff], 0, 0);
      __builtin_amdgcn_global_load_async_to_lds_b128((AS1 v4i_t*)gal, (AS3 v4i_t*)&shAl[buf][loff], 0, 0);
      __builtin_amdgcn_global_load_async_to_lds_b128((AS1 v4i_t*)gbh, (AS3 v4i_t*)&shBh[buf][loff], 0, 0);
      __builtin_amdgcn_global_load_async_to_lds_b128((AS1 v4i_t*)gbl, (AS3 v4i_t*)&shBl[buf][loff], 0, 0);
#else
      *(uint4*)&shAh[buf][loff] = *(const uint4*)gah;
      *(uint4*)&shAl[buf][loff] = *(const uint4*)gal;
      *(uint4*)&shBh[buf][loff] = *(const uint4*)gbh;
      *(uint4*)&shBl[buf][loff] = *(const uint4*)gbl;
#endif
    }
  };

  const int KT = GN / KC;  // 64
  stage(0, 0);

  for (int kt = 0; kt < KT; ++kt) {
    const int buf = kt & 1;
    if (kt + 1 < KT) {
      stage(buf ^ 1, kt + 1);
      WAIT_ASYNC(8);   // 8 ops of the next stage still outstanding
    } else {
      WAIT_ASYNC(0);
    }
    __syncthreads();

    // ---- A fragments: ISA 16x32 bf16 layout. VGPR p holds K pair
    //      k = 2p + (p>=4 ? 8 : 0) + 8*half  -> contiguous bf16 pair. ----
    frag_u a_h[2], a_l[2];
    #pragma unroll
    for (int mt = 0; mt < 2; ++mt) {
      const int base = (wm + mt * 16 + l16) * LDT + (hlf << 3);
      #pragma unroll
      for (int p = 0; p < 8; ++p) {
        const int off = base + 2 * p + ((p >> 2) << 3);
        a_h[mt].w[p] = *(const uint32_t*)&shAh[buf][off];
        a_l[mt].w[p] = *(const uint32_t*)&shAl[buf][off];
      }
    }

    // ---- B fragments: VGPR p holds K pair k = 2p + (p>=4 ? 8 : 0) + 8*half
    //      (column n of effective B = row n of K-major tile) ----
    #pragma unroll
    for (int nt = 0; nt < 4; ++nt) {
      frag_u b_h, b_l;
      const int base = (wn + nt * 16 + l16) * LDT + (hlf << 3);
      #pragma unroll
      for (int p = 0; p < 8; ++p) {
        const int off = base + 2 * p + ((p >> 2) << 3);
        b_h.w[p] = *(const uint32_t*)&shBh[buf][off];
        b_l.w[p] = *(const uint32_t*)&shBl[buf][off];
      }
      #pragma unroll
      for (int mt = 0; mt < 2; ++mt) {
        acc[mt][nt] = __builtin_amdgcn_wmma_f32_16x16x32_bf16(
            false, a_h[mt].v, false, b_h.v, (short)0, acc[mt][nt], false, false);
        acc[mt][nt] = __builtin_amdgcn_wmma_f32_16x16x32_bf16(
            false, a_h[mt].v, false, b_l.v, (short)0, acc[mt][nt], false, false);
        acc[mt][nt] = __builtin_amdgcn_wmma_f32_16x16x32_bf16(
            false, a_l[mt].v, false, b_h.v, (short)0, acc[mt][nt], false, false);
      }
    }
    __syncthreads();
  }

  // ---- epilogue: C/D layout (VGPR r: m = r + 8*half, n = lane&15) ----
  #pragma unroll
  for (int mt = 0; mt < 2; ++mt)
    #pragma unroll
    for (int nt = 0; nt < 4; ++nt) {
      const int col = colBase + wn + nt * 16 + l16;
      #pragma unroll
      for (int r = 0; r < 8; ++r) {
        const int row = rowBase + wm + mt * 16 + r + hlf * 8;
        __bf16 h, l;
        split2(acc[mt][nt][r], h, l);
        Ch[(size_t)row * GN + col] = h;
        Cl[(size_t)row * GN + col] = l;
        if (Cth) {
          Cth[(size_t)col * GN + row] = h;
          Ctl[(size_t)col * GN + row] = l;
        }
      }
    }
}

// ============================================================================
// Phase-1 reductions and elementwise kernels
// ============================================================================
__global__ __launch_bounds__(256) void k_row_reduce(const float* __restrict__ cov,
                                                    const float* __restrict__ mu,
                                                    float* var, float* rowsq, float* sigmu) {
  const int i = blockIdx.x;
  const float* row = cov + (size_t)i * GN;
  float rs = 0.f, sm = 0.f;
  for (int j = threadIdx.x; j < GN; j += 256) {
    const float c = row[j];
    rs += c * c;
    sm += c * mu[j];
  }
  const float a = block_sum(rs);
  const float b = block_sum(sm);
  if (threadIdx.x == 0) {
    rowsq[i] = a;
    sigmu[i] = b;
    var[i]   = row[i];
  }
}

// scal: 0=frob2 1=total_x2 2=quad 3=mse 4=tr_c1 5=tr_c2 6=trP 7=trY
__global__ __launch_bounds__(256) void k_sum3(const float* mu, const float* var,
                                              const float* rowsq, const float* sigmu,
                                              float* scal) {
  float fro = 0.f, tot = 0.f, qd = 0.f;
  for (int i = threadIdx.x; i < GN; i += 256) {
    const float m = mu[i];
    fro += rowsq[i];
    tot += m * m + var[i];
    qd  += m * sigmu[i];
  }
  const float a = block_sum(fro);
  const float b = block_sum(tot);
  const float c = block_sum(qd);
  if (threadIdx.x == 0) { scal[0] = a; scal[1] = b; scal[2] = c; }
}

__global__ __launch_bounds__(256) void k_mu_y(const float* mu, const float* mu_obs,
                                              const float* var, const float* rowsq,
                                              const float* sigmu, const float* scal,
                                              float* muy, float* tvec) {
  const int i = blockIdx.x * 256 + threadIdx.x;
  const float frob2 = scal[0], total = scal[1], quad = scal[2];
  const float u = mu[i], vi = var[i], smi = sigmu[i];
  const float mu2 = u * u;
  const float v_mean = total - mu2 - vi;
  const float v_var = 2.f * (frob2 - 2.f * rowsq[i] + vi * vi) +
                      4.f * (quad - 2.f * u * smi + mu2 * vi);
  const float v_cov = 2.f * (smi - vi * u);
  const float w = mu2 + v_mean;
  const float w05 = 1.f / sqrtf(w);
  const float w15 = w05 / w;
  const float w25 = w15 / w;
  const float f0 = u * w05;
  const float fuu = -3.f * u * v_mean * w25;
  const float fvv = 0.75f * u * w25;
  const float fuv = w15 - 1.5f * v_mean * w25;
  const float y = f0 + 0.5f * fuu * vi + 0.5f * fvv * v_var + fuv * v_cov;
  muy[i] = y;
  const float d = y - mu_obs[i];
  tvec[i] = d * d;
}

__global__ __launch_bounds__(256) void k_sum1(const float* v, float* slot) {
  float a = 0.f;
  for (int i = threadIdx.x; i < GN; i += 256) a += v[i];
  const float s = block_sum(a);
  if (threadIdx.x == 0) *slot = s;
}

__global__ __launch_bounds__(256) void k_split(const float* __restrict__ src,
                                               __bf16* __restrict__ h,
                                               __bf16* __restrict__ l) {
  const int idx = blockIdx.x * 256 + threadIdx.x;
  split2(src[idx], h[idx], l[idx]);
}

__global__ __launch_bounds__(256) void k_build_c1(const float* __restrict__ cov,
                                                  const __bf16* __restrict__ S2h,
                                                  const __bf16* __restrict__ S2l,
                                                  const float* mu, const float* sigmu,
                                                  const float* muy, const float* scal,
                                                  __bf16* __restrict__ c1h,
                                                  __bf16* __restrict__ c1l) {
  const int idx = blockIdx.x * 256 + threadIdx.x;
  const int i = idx >> 11, j = idx & (GN - 1);
  const float md = scal[1];
  const float vd = 2.f * scal[0] + 4.f * scal[2];
  const float mn = cov[idx] + mu[i] * mu[j];
  const float cnd = 2.f * up2(S2h[idx], S2l[idx]) +
                    2.f * (mu[i] * sigmu[j] + sigmu[i] * mu[j]);
  const float inv = 1.f / md;
  const float sm = mn * inv - cnd * inv * inv + mn * (vd * inv * inv * inv);
  split2(sm - muy[i] * muy[j], c1h[idx], c1l[idx]);
}

__global__ __launch_bounds__(256) void k_diag_pl(const __bf16* h, const __bf16* l,
                                                 float* slot) {
  float a = 0.f;
  for (int i = threadIdx.x; i < GN; i += 256) {
    const size_t d = (size_t)i * GN + i;
    a += up2(h[d], l[d]);
  }
  const float s = block_sum(a);
  if (threadIdx.x == 0) *slot = s;
}

__global__ __launch_bounds__(256) void k_diag_f32(const float* M, float* slot) {
  float a = 0.f;
  for (int i = threadIdx.x; i < GN; i += 256) a += M[(size_t)i * GN + i];
  const float s = block_sum(a);
  if (threadIdx.x == 0) *slot = s;
}

__global__ __launch_bounds__(256) void k_scale_pl(const __bf16* __restrict__ Ph,
                                                  const __bf16* __restrict__ Pl,
                                                  const float* s_slot,
                                                  __bf16* __restrict__ Yh,
                                                  __bf16* __restrict__ Yl,
                                                  __bf16* __restrict__ Yth,
                                                  __bf16* __restrict__ Ytl) {
  const int idx = blockIdx.x * 256 + threadIdx.x;
  const int i = idx >> 11, j = idx & (GN - 1);
  __bf16 h, l;
  split2(up2(Ph[idx], Pl[idx]) / s_slot[0], h, l);
  Yh[idx] = h;
  Yl[idx] = l;
  const size_t tr = (size_t)j * GN + i;
  Yth[tr] = h;
  Ytl[tr] = l;
}

__global__ __launch_bounds__(256) void k_identity_pl(__bf16* __restrict__ h,
                                                     __bf16* __restrict__ l) {
  const int idx = blockIdx.x * 256 + threadIdx.x;
  const int i = idx >> 11, j = idx & (GN - 1);
  h[idx] = bf_bits((i == j) ? 0x3F80u : 0u);
  l[idx] = bf_bits(0u);
}

__global__ __launch_bounds__(256) void k_ns_w_pl(const __bf16* __restrict__ Th,
                                                 const __bf16* __restrict__ Tl,
                                                 __bf16* __restrict__ Wh,
                                                 __bf16* __restrict__ Wl,
                                                 __bf16* __restrict__ Wth,
                                                 __bf16* __restrict__ Wtl) {
  const int idx = blockIdx.x * 256 + threadIdx.x;
  const int i = idx >> 11, j = idx & (GN - 1);
  const float w = ((i == j) ? 1.5f : 0.f) - 0.5f * up2(Th[idx], Tl[idx]);
  __bf16 h, l;
  split2(w, h, l);
  Wh[idx] = h;
  Wl[idx] = l;
  const size_t tr = (size_t)j * GN + i;
  Wth[tr] = h;
  Wtl[tr] = l;
}

__global__ void k_finalize(const float* scal, float* out) {
  if (threadIdx.x == 0 && blockIdx.x == 0)
    out[0] = scal[3] + scal[4] + scal[5] -
             2.f * sqrtf(fmaxf(scal[6], 0.f)) * scal[7];
}

// ============================================================================
// Host orchestration
// ============================================================================
extern "C" void kernel_launch(void* const* d_in, const int* in_sizes, int n_in,
                              void* d_out, int out_size, void* d_ws, size_t ws_size,
                              hipStream_t stream) {
  (void)in_sizes; (void)n_in; (void)out_size; (void)ws_size;
  const float* mu      = (const float*)d_in[0];
  const float* cov     = (const float*)d_in[1];
  const float* mu_obs  = (const float*)d_in[2];
  const float* cov_obs = (const float*)d_in[3];
  float* out = (float*)d_out;

  char* wsb = (char*)d_ws;
  float* scal  = (float*)wsb;               // 8 scalar slots
  float* var   = (float*)(wsb + 4096);
  float* rowsq = var + GN;
  float* sigmu = rowsq + GN;
  float* muy   = sigmu + GN;
  float* tvec  = muy + GN;
  // bf16 planes, 8 MB each (22 planes used, ~176 MB)
  auto PL = [&](int i) {
    return (__bf16*)(wsb + 65536 + (size_t)i * (size_t)NNE * 2u);
  };

  const dim3 gemm_grid(GN / BLKT, GN / BLKT);
  const int EW = NNE / 256;  // elementwise blocks

  auto gemm = [&](const __bf16* Ah, const __bf16* Al,
                  const __bf16* Bth, const __bf16* Btl,
                  __bf16* Ch, __bf16* Cl, __bf16* Cth, __bf16* Ctl) {
    k_gemm_nt<<<gemm_grid, 256, 0, stream>>>(Ah, Al, Bth, Btl, Ch, Cl, Cth, Ctl);
  };

  // --- Taylor moments ---
  k_row_reduce<<<GN, 256, 0, stream>>>(cov, mu, var, rowsq, sigmu);
  k_sum3<<<1, 256, 0, stream>>>(mu, var, rowsq, sigmu, scal);
  k_mu_y<<<GN / 256, 256, 0, stream>>>(mu, mu_obs, var, rowsq, sigmu, scal, muy, tvec);
  k_sum1<<<1, 256, 0, stream>>>(tvec, scal + 3);

  k_split<<<EW, 256, 0, stream>>>(cov, PL(0), PL(1));                    // cov (symmetric)
  gemm(PL(0), PL(1), PL(0), PL(1), PL(2), PL(3), nullptr, nullptr);      // S2 = cov@cov
  k_build_c1<<<EW, 256, 0, stream>>>(cov, PL(2), PL(3), mu, sigmu, muy, scal,
                                     PL(4), PL(5));                      // c1 (symmetric)
  k_diag_pl<<<1, 256, 0, stream>>>(PL(4), PL(5), scal + 4);              // tr(c1)
  k_split<<<EW, 256, 0, stream>>>(cov_obs, PL(6), PL(7));                // c2 (symmetric)
  k_diag_f32<<<1, 256, 0, stream>>>(cov_obs, scal + 5);                  // tr(c2)

  // --- Newton-Schulz for tr(sqrtm(c1@c2)) ---
  gemm(PL(4), PL(5), PL(6), PL(7), PL(8), PL(9), nullptr, nullptr);      // P = c1@c2
  k_diag_pl<<<1, 256, 0, stream>>>(PL(8), PL(9), scal + 6);              // s = tr(P)
  // Y0 = P/s  (planes 0-3 free now: h, l, th, tl)
  k_scale_pl<<<EW, 256, 0, stream>>>(PL(8), PL(9), scal + 6,
                                     PL(0), PL(1), PL(2), PL(3));
  // Z0 = I (planes 4-7 free; identity is symmetric so th/tl same content)
  k_identity_pl<<<EW, 256, 0, stream>>>(PL(4), PL(5));
  k_identity_pl<<<EW, 256, 0, stream>>>(PL(6), PL(7));

  __bf16 *Yh = PL(0),  *Yl = PL(1),  *Yth = PL(2),  *Ytl = PL(3);
  __bf16 *Zh = PL(4),  *Zl = PL(5),  *Zth = PL(6),  *Ztl = PL(7);
  __bf16 *Th = PL(8),  *Tl = PL(9);                                      // reuse P
  __bf16 *Wh = PL(10), *Wl = PL(11), *Wth = PL(12), *Wtl = PL(13);
  __bf16 *Y2h = PL(14), *Y2l = PL(15), *Y2th = PL(16), *Y2tl = PL(17);
  __bf16 *Z2h = PL(18), *Z2l = PL(19), *Z2th = PL(20), *Z2tl = PL(21);

  for (int it = 0; it < NS_ITERS; ++it) {
    gemm(Zh, Zl, Yth, Ytl, Th, Tl, nullptr, nullptr);         // T = Z@Y
    k_ns_w_pl<<<EW, 256, 0, stream>>>(Th, Tl, Wh, Wl, Wth, Wtl);  // W = 1.5I-0.5T
    gemm(Yh, Yl, Wth, Wtl, Y2h, Y2l, Y2th, Y2tl);             // Y' = Y@W
    gemm(Wh, Wl, Zth, Ztl, Z2h, Z2l, Z2th, Z2tl);             // Z' = W@Z
    __bf16* tp;
    tp = Yh; Yh = Y2h; Y2h = tp;   tp = Yl; Yl = Y2l; Y2l = tp;
    tp = Yth; Yth = Y2th; Y2th = tp; tp = Ytl; Ytl = Y2tl; Y2tl = tp;
    tp = Zh; Zh = Z2h; Z2h = tp;   tp = Zl; Zl = Z2l; Z2l = tp;
    tp = Zth; Zth = Z2th; Z2th = tp; tp = Ztl; Ztl = Z2tl; Z2tl = tp;
  }
  k_diag_pl<<<1, 256, 0, stream>>>(Yh, Yl, scal + 7);          // tr(Y_final)
  k_finalize<<<1, 1, 0, stream>>>(scal, out);
}
